// MyRNN_37280316129803
// MI455X (gfx1250) — matrix operand order repaired and checked by reading, not compiled, not run on hardware
//
#include <hip/hip_runtime.h>
#include <math.h>

typedef __attribute__((ext_vector_type(16))) _Float16 v16h;
typedef __attribute__((ext_vector_type(8)))  _Float16 v8h;
typedef __attribute__((ext_vector_type(8)))  float    v8f;
typedef __attribute__((ext_vector_type(4)))  float    v4f;

#define BB 64
#define SS 512
#define DD 1024
#define HH 1024
#define NBLK 32          // blocks in the persistent recurrence kernel

// ---------------------------------------------------------------------------
// wt[c][r] = (f16) w[r][c]   for a 1024x1024 f32 matrix
// ---------------------------------------------------------------------------
__global__ __launch_bounds__(256) void k_transpose_f16(
    const float* __restrict__ w, _Float16* __restrict__ wt) {
  int idx = blockIdx.x * 256 + threadIdx.x;      // 0 .. 1M-1
  int r = idx >> 10;
  int c = idx & 1023;
  wt[(size_t)c * DD + r] = (_Float16)w[idx];
}

__global__ __launch_bounds__(256) void k_zero_u32(uint32_t* __restrict__ p, int n) {
  int i = blockIdx.x * 256 + threadIdx.x;
  if (i < n) p[i] = 0u;
}

// ---------------------------------------------------------------------------
// x_proj: out[r][h] = sum_d x[r][d] * Wxh[d][h] + b[h],  r = b*S + s
// one wave -> 16x64 output strip (4 accumulators), K-loop of WMMA 16x16x32
// ---------------------------------------------------------------------------
__global__ __launch_bounds__(256) void k_xproj_wmma(
    const float* __restrict__ x,        // [32768 x 1024] f32
    const _Float16* __restrict__ wt,    // [HH][DD] = Wxh^T f16
    const float* __restrict__ bias,     // [HH] f32
    float* __restrict__ out)            // [32768 x 1024] f32
{
  const int lane = threadIdx.x & 31;
  const int wave = (blockIdx.x * 256 + threadIdx.x) >> 5;
  const int NG   = HH / 64;             // 16 n-groups of 64 cols
  const int mt   = wave / NG;           // 0..2047
  const int ng   = wave % NG;
  const int m    = lane & 15;
  const int half = lane >> 4;
  const int rowA = mt * 16 + m;
  const int ncol = ng * 64 + (lane & 15);

  v8f acc[4] = {};
  const float* arow = x + (size_t)rowA * DD;

  for (int kt = 0; kt < DD; kt += 32) {
    v4f f0 = *(const v4f*)(arow + kt + half * 8);
    v4f f1 = *(const v4f*)(arow + kt + half * 8 + 4);
    v4f f2 = *(const v4f*)(arow + kt + 16 + half * 8);
    v4f f3 = *(const v4f*)(arow + kt + 16 + half * 8 + 4);
    v16h a;
#pragma unroll
    for (int i = 0; i < 4; ++i) {
      a[i]      = (_Float16)f0[i];
      a[4 + i]  = (_Float16)f1[i];
      a[8 + i]  = (_Float16)f2[i];
      a[12 + i] = (_Float16)f3[i];
    }
#pragma unroll
    for (int j = 0; j < 4; ++j) {
      const _Float16* bp = wt + (size_t)(ncol + j * 16) * DD + kt + half * 8;
      v8h blo = *(const v8h*)bp;
      v8h bhi = *(const v8h*)(bp + 16);
      v16h b;
#pragma unroll
      for (int i = 0; i < 8; ++i) { b[i] = blo[i]; b[8 + i] = bhi[i]; }
      acc[j] = __builtin_amdgcn_wmma_f32_16x16x32_f16(
          false, a, false, b, (short)0, acc[j], false, false);
    }
  }

#pragma unroll
  for (int j = 0; j < 4; ++j) {
    const int col = ncol + j * 16;
    const float bv = bias[col];
#pragma unroll
    for (int r = 0; r < 8; ++r) {
      const int mrow = half * 8 + r;
      out[(size_t)(mt * 16 + mrow) * HH + col] = acc[j][r] + bv;
    }
  }
}

// ---------------------------------------------------------------------------
// Persistent recurrence kernel: all 512 steps in one launch.
//   32 blocks x 256 threads; block b owns output columns [b*32, b*32+32):
//     wave w: mt = w&3 (batch tile), ntl = w>>2 -> one 16x16 tile per wave.
//   Whh^T columns staged ONCE into LDS via global_load_async_to_lds_b128
//   (ASYNCcnt path), XOR-swizzled 16B chunks -> conflict-free ds_load_b128.
//   k-loop: A (h_prev, global/L2) prefetched at distance 4; B distance 1
//   from LDS; epilogue xp values loaded before the k-loop.
//   Grid-wide step barrier: per-thread release fence -> relaxed atomic
//   counter -> spin w/ s_sleep -> per-thread acquire fence.
// ---------------------------------------------------------------------------
__global__ __launch_bounds__(256) void k_rnn_persistent(
    const _Float16* __restrict__ whht,  // [HH][HH] = Whh^T f16
    float* out,                         // [B][S][H] f32 (x_proj in, h out)
    _Float16* h0,                       // [64][1024] f16 (zeroed)
    _Float16* h1,                       // [64][1024] f16
    unsigned int* bar)                  // zeroed barrier counter
{
  __shared__ __align__(16) _Float16 sB[32 * 1024];   // 64 KB

  const int colbase = blockIdx.x * 32;
  // ---- stage this block's 32 columns of Whh^T into LDS, async DMA path
  for (int t = threadIdx.x; t < 32 * 128; t += 256) {
    const int c  = t >> 7;        // local column 0..31
    const int ch = t & 127;       // 16B chunk within the 1024-half column
    unsigned lds = (unsigned)(uintptr_t)(sB + c * 1024 + ((ch ^ (c & 15)) << 3));
    unsigned long long ga =
        (unsigned long long)(uintptr_t)(whht + (size_t)(colbase + c) * HH + (ch << 3));
    asm volatile("global_load_async_to_lds_b128 %0, %1, off"
                 :: "v"(lds), "v"(ga) : "memory");
  }
  asm volatile("s_wait_asynccnt 0x0" ::: "memory");
  __syncthreads();

  const int lane = threadIdx.x & 31;
  const int wave = threadIdx.x >> 5;    // 0..7
  const int mt   = wave & 3;            // batch tile 0..3
  const int ntl  = wave >> 2;           // local n tile 0..1
  const int nt   = blockIdx.x * 2 + ntl;
  const int half = lane >> 4;
  const int l16  = lane & 15;
  const int c    = ntl * 16 + l16;      // local column for B fragment
  const int cx   = c & 15;
  const _Float16* bbase = sB + c * 1024;
  const int col  = nt * 16 + l16;       // global output column

  const _Float16* hp = h0;
  _Float16* hn = h1;
  unsigned int target = 0;

  for (int s = 0; s < SS; ++s) {
    const _Float16* arow = hp + (size_t)(mt * 16 + l16) * HH;

    // epilogue inputs issued early: latency hides under the whole k-loop
    float xp[8];
#pragma unroll
    for (int r = 0; r < 8; ++r)
      xp[r] = out[((size_t)(mt * 16 + half * 8 + r) * SS + s) * HH + col];

    v8f acc = {};

    // A prefetch pipeline, depth 4 (global/L2); B depth 1 (LDS)
    v8h aL[4], aH[4];
#pragma unroll
    for (int p = 0; p < 4; ++p) {
      aL[p] = *(const v8h*)(arow + p * 32 + half * 8);
      aH[p] = *(const v8h*)(arow + p * 32 + 16 + half * 8);
    }
    const int ch0 = half;
    v8h bL = *(const v8h*)(bbase + ((ch0 ^ cx) << 3));
    v8h bH = *(const v8h*)(bbase + (((ch0 + 2) ^ cx) << 3));

#pragma unroll
    for (int kt = 0; kt < HH; kt += 32) {
      const int slot = (kt >> 5) & 3;
      v16h a, b;
#pragma unroll
      for (int i = 0; i < 8; ++i) {
        a[i] = aL[slot][i]; a[8 + i] = aH[slot][i];
        b[i] = bL[i];       b[8 + i] = bH[i];
      }
      if (kt + 128 < HH) {
        aL[slot] = *(const v8h*)(arow + kt + 128 + half * 8);
        aH[slot] = *(const v8h*)(arow + kt + 128 + 16 + half * 8);
      }
      if (kt + 32 < HH) {
        const int ch = ((kt + 32) >> 3) + half;
        bL = *(const v8h*)(bbase + ((ch ^ cx) << 3));
        bH = *(const v8h*)(bbase + (((ch + 2) ^ cx) << 3));
      }
      acc = __builtin_amdgcn_wmma_f32_16x16x32_f16(
          false, a, false, b, (short)0, acc, false, false);
    }

    // epilogue: h_new = tanh(xp + acc)
#pragma unroll
    for (int r = 0; r < 8; ++r) {
      const int bb = mt * 16 + half * 8 + r;
      const size_t o = ((size_t)bb * SS + s) * HH + col;
      const float v = tanhf(acc[r] + xp[r]);
      out[o] = v;
      hn[(size_t)bb * HH + col] = (_Float16)v;
    }

    // ---- device-wide step barrier (monotonic counter)
    __builtin_amdgcn_fence(__ATOMIC_RELEASE, "agent");   // every thread flushes
    __syncthreads();
    target += NBLK;
    if (threadIdx.x == 0) {
      __hip_atomic_fetch_add(bar, 1u, __ATOMIC_RELAXED, __HIP_MEMORY_SCOPE_AGENT);
      while (__hip_atomic_load(bar, __ATOMIC_RELAXED, __HIP_MEMORY_SCOPE_AGENT) < target)
        __builtin_amdgcn_s_sleep(1);
    }
    __syncthreads();
    __builtin_amdgcn_fence(__ATOMIC_ACQUIRE, "agent");   // every thread invalidates

    _Float16* tmp = (_Float16*)hp; hp = hn; hn = tmp;
  }
}

// ---------------------------------------------------------------------------
extern "C" void kernel_launch(void* const* d_in, const int* in_sizes, int n_in,
                              void* d_out, int out_size, void* d_ws, size_t ws_size,
                              hipStream_t stream) {
  const float* x_seq = (const float*)d_in[0];   // [B,S,D] f32
  const float* w_xh  = (const float*)d_in[1];   // [D,H]  f32
  const float* w_hh  = (const float*)d_in[2];   // [H,H]  f32
  const float* b_h   = (const float*)d_in[3];   // [H]    f32
  float* out = (float*)d_out;                   // [B,S,H] f32

  _Float16* wxht = (_Float16*)d_ws;                     // 2 MB
  _Float16* whht = wxht + (size_t)DD * HH;              // 2 MB
  _Float16* h0   = whht + (size_t)HH * HH;              // 128 KB
  _Float16* h1   = h0 + (size_t)BB * HH;                // 128 KB
  unsigned int* bar = (unsigned int*)(h1 + (size_t)BB * HH);

  // weight transpose + f16 convert (B-fragments become contiguous loads)
  k_transpose_f16<<<(DD * HH) / 256, 256, 0, stream>>>(w_xh, wxht);
  k_transpose_f16<<<(HH * HH) / 256, 256, 0, stream>>>(w_hh, whht);
  // h(0) = 0 and barrier counter = 0
  k_zero_u32<<<(BB * HH / 2) / 256, 256, 0, stream>>>((uint32_t*)h0, BB * HH / 2);
  k_zero_u32<<<1, 256, 0, stream>>>((uint32_t*)bar, 1);

  // x_proj for all timesteps, written into d_out in its final [B,S,H] layout
  {
    const int waves = (BB * SS / 16) * (HH / 64);       // 32768 waves
    k_xproj_wmma<<<waves / 8, 256, 0, stream>>>(x_seq, wxht, b_h, out);
  }

  // all 512 dependent steps in one persistent cooperative kernel
  k_rnn_persistent<<<NBLK, 256, 0, stream>>>(whht, out, h0, h1, bar);
}
